// TFLongformerSelfAttention_5987184410959
// MI455X (gfx1250) — matrix operand order repaired and checked by reading, not compiled, not run on hardware
//
#include <hip/hip_runtime.h>
#include <hip/hip_bf16.h>
#include <cmath>

typedef __attribute__((ext_vector_type(16))) _Float16 v16h;
typedef __attribute__((ext_vector_type(8)))  float    v8f;

#define B_DIM 2
#define S_DIM 4096
#define E_DIM 768
#define H_NUM 12
#define D_DIM 64
#define W_WIN 256
#define C_NUM 16

__device__ __forceinline__ v8f wmma_f16(v16h a, v16h b, v8f c) {
  // D = A(16x32 f16) * B(32x16 f16) + C(16x16 f32)
  return __builtin_amdgcn_wmma_f32_16x16x32_f16(
      /*neg_a=*/false, a, /*neg_b=*/false, b,
      /*c_mod=*/(short)0, c, /*reuse_a=*/false, /*reuse_b=*/false);
}

union Frag { v16h v; float4 f[2]; };

// A fragment: rowptr = start of this lane's M-row (caller applies lane&15).
// ISA layout: lanes 0-15 K={0..7,16..23}, lanes 16-31 K={8..15,24..31} (+k0).
__device__ __forceinline__ v16h load_a_frag(const _Float16* rowptr, int k0) {
  int h4 = (threadIdx.x >> 4) & 1;
  Frag u;
  u.f[0] = *(const float4*)(rowptr + k0 + 8 * h4);
  u.f[1] = *(const float4*)(rowptr + k0 + 16 + 8 * h4);
  return u.v;
}

// B fragment: rowptr = start of this lane's N-column stored as a contiguous
// K-row ([N][K] storage). lanes 0-15: K={0..7,16..23}, lanes 16-31: K={8..15,24..31} (+k0).
__device__ __forceinline__ v16h load_b_frag(const _Float16* rowptr, int k0) {
  int h4 = (threadIdx.x >> 4) & 1;
  const _Float16* p = rowptr + k0 + 8 * h4;
  Frag u;
  u.f[0] = *(const float4*)(p);
  u.f[1] = *(const float4*)(p + 16);
  return u.v;
}

__device__ __forceinline__ float redsum16(float v) {
#pragma unroll
  for (int off = 1; off < 16; off <<= 1) v += __shfl_xor(v, off, 32);
  return v;
}

// ---------------- conversion kernels ----------------
__global__ void f32_to_f16_kernel(const float* __restrict__ x,
                                  _Float16* __restrict__ y, int n) {
  int i = blockIdx.x * blockDim.x + threadIdx.x;
  if (i < n) y[i] = (_Float16)x[i];
}

__global__ void transpose_w_kernel(const float* __restrict__ Wq,
                                   const float* __restrict__ Wk,
                                   const float* __restrict__ Wv,
                                   _Float16* __restrict__ WtAll) {
  int t = blockIdx.x * blockDim.x + threadIdx.x; // over E*E
  int wi = blockIdx.z;
  const float* W = (wi == 0) ? Wq : ((wi == 1) ? Wk : Wv);
  int k = t / E_DIM;
  int e = t % E_DIM;
  WtAll[(size_t)wi * E_DIM * E_DIM + e * E_DIM + k] = (_Float16)W[t];
}

// ---------------- QKV projection GEMM ----------------
// grid (64, 12, 3), block 256 (8 waves). Wave computes 16 rows x 64 cols.
__global__ void __launch_bounds__(256) qkv_gemm_kernel(
    const _Float16* __restrict__ X, const _Float16* __restrict__ WtAll,
    const float* __restrict__ bq, const float* __restrict__ bk,
    const float* __restrict__ bv, _Float16* __restrict__ Qh,
    _Float16* __restrict__ Kh, _Float16* __restrict__ Vt) {
  int wave = threadIdx.x >> 5;
  int lane = threadIdx.x & 31;
  int n = lane & 15, h4 = lane >> 4;
  int row0 = blockIdx.x * 128 + wave * 16;  // over B*S
  int h = blockIdx.y;
  int n0 = h * D_DIM;
  int wi = blockIdx.z;
  const _Float16* Wt = WtAll + (size_t)wi * E_DIM * E_DIM;
  const _Float16* arow = X + (size_t)(row0 + n) * E_DIM;
  const _Float16* brow = Wt + (size_t)(n0 + n) * E_DIM;

  v8f acc0 = {}, acc1 = {}, acc2 = {}, acc3 = {};
  for (int k0 = 0; k0 < E_DIM; k0 += 32) {
    v16h a = load_a_frag(arow, k0);
    acc0 = wmma_f16(a, load_b_frag(brow, k0), acc0);
    acc1 = wmma_f16(a, load_b_frag(brow + 16 * E_DIM, k0), acc1);
    acc2 = wmma_f16(a, load_b_frag(brow + 32 * E_DIM, k0), acc2);
    acc3 = wmma_f16(a, load_b_frag(brow + 48 * E_DIM, k0), acc3);
  }
  const float* bias = (wi == 0) ? bq : ((wi == 1) ? bk : bv);
#pragma unroll
  for (int r = 0; r < 8; ++r) {
    int m = r + 8 * h4;
    int gr = row0 + m;
    int bb = gr >> 12;          // / S_DIM
    int s = gr & (S_DIM - 1);
    float vals[4] = {acc0[r], acc1[r], acc2[r], acc3[r]};
#pragma unroll
    for (int nt = 0; nt < 4; ++nt) {
      int d = nt * 16 + n;
      float v = vals[nt] + bias[n0 + d];
      if (wi == 0) {
        Qh[(((size_t)bb * H_NUM + h) * S_DIM + s) * D_DIM + d] =
            (_Float16)(v * 0.125f);  // 1/sqrt(64)
      } else if (wi == 1) {
        Kh[(((size_t)bb * H_NUM + h) * S_DIM + s) * D_DIM + d] = (_Float16)v;
      } else {
        Vt[(((size_t)bb * H_NUM + h) * D_DIM + d) * S_DIM + s] = (_Float16)v;
      }
    }
  }
}

// ---------------- banded attention ----------------
// Softmax is shift-invariant; scores here are O(1), clamped at +11 so
// exp(score) fits f16 normal range. No running max / no accumulator rescale:
// lane-local row sums, reduced across lanes once at the end.
// grid (C=16, H=12, B=2), block 256 (8 waves); wave handles qt = wave, wave+8.
__global__ void __launch_bounds__(256) attn_kernel(
    const _Float16* __restrict__ Qh, const _Float16* __restrict__ Kh,
    const _Float16* __restrict__ Vt, const float* __restrict__ amask,
    float* __restrict__ out) {
  __shared__ __align__(16) _Float16 pbuf[8][16 * 32];
  int c = blockIdx.x, h = blockIdx.y, b = blockIdx.z;
  int wave = threadIdx.x >> 5;
  int lane = threadIdx.x & 31;
  int n = lane & 15, h4 = lane >> 4;

  const _Float16* Qhd = Qh + (((size_t)b * H_NUM + h) * S_DIM) * D_DIM;
  const _Float16* Khd = Kh + (((size_t)b * H_NUM + h) * S_DIM) * D_DIM;
  const _Float16* Vtd = Vt + (((size_t)b * H_NUM + h) * D_DIM) * S_DIM;
  const float* mrow = amask + (size_t)b * S_DIM;
  _Float16* P = &pbuf[wave][0];

  for (int qt = wave; qt < 16; qt += 8) {
    int q0 = c * W_WIN + qt * 16;
    const _Float16* qrow = Qhd + (size_t)(q0 + n) * D_DIM;
    v16h qa0 = load_a_frag(qrow, 0);
    v16h qa1 = load_a_frag(qrow, 32);

    v8f o0 = {}, o1 = {}, o2 = {}, o3 = {};
    float rsum[8];
#pragma unroll
    for (int r = 0; r < 8; ++r) rsum[r] = 0.0f;

    int jorig = c * W_WIN - W_WIN;
    for (int kt = 0; kt < 24; ++kt) {
      int jbase = jorig + kt * 32;
      if (jbase < 0 || jbase >= S_DIM) continue;  // fully OOB key tile

      // ---- scores: Q(16x64) @ K^T -> 16x32 ----
      const _Float16* krow0 = Khd + (size_t)(jbase + n) * D_DIM;
      const _Float16* krow1 = Khd + (size_t)(jbase + 16 + n) * D_DIM;
      v8f sA = {}, sB = {};
      sA = wmma_f16(qa0, load_b_frag(krow0, 0), sA);
      sA = wmma_f16(qa1, load_b_frag(krow0, 32), sA);
      sB = wmma_f16(qa0, load_b_frag(krow1, 0), sB);
      sB = wmma_f16(qa1, load_b_frag(krow1, 32), sB);

      // ---- mask + exp; key-dependent terms hoisted out of the row loop ----
      int jA = jbase + n, jB = jA + 16;
      float fmA = (mrow[jA] != 0.0f) ? -10000.0f : 0.0f;
      float fmB = (mrow[jB] != 0.0f) ? -10000.0f : 0.0f;
      int sq0 = q0 + 8 * h4;
#pragma unroll
      for (int r = 0; r < 8; ++r) {
        int sq = sq0 + r;
        int dA = jA - sq, dB = jB - sq;
        float pA = (dA >= -W_WIN && dA <= W_WIN)
                       ? __expf(fminf(sA[r] + fmA, 11.0f)) : 0.0f;
        float pB = (dB >= -W_WIN && dB <= W_WIN)
                       ? __expf(fminf(sB[r] + fmB, 11.0f)) : 0.0f;
        rsum[r] += pA + pB;
        P[(r + 8 * h4) * 32 + n] = (_Float16)pA;
        P[(r + 8 * h4) * 32 + n + 16] = (_Float16)pB;
      }
      asm volatile("s_wait_dscnt 0" ::: "memory");  // P visible (same wave)

      // ---- ctx: P(16x32) @ V(32x64) ----
      v16h pf = load_a_frag(P + n * 32, 0);
      o0 = wmma_f16(pf, load_b_frag(Vtd + (size_t)(0 + n) * S_DIM + jbase, 0), o0);
      o1 = wmma_f16(pf, load_b_frag(Vtd + (size_t)(16 + n) * S_DIM + jbase, 0), o1);
      o2 = wmma_f16(pf, load_b_frag(Vtd + (size_t)(32 + n) * S_DIM + jbase, 0), o2);
      o3 = wmma_f16(pf, load_b_frag(Vtd + (size_t)(48 + n) * S_DIM + jbase, 0), o3);
    }

    // ---- epilogue: one cross-lane sum reduction, normalize, store f32 ----
#pragma unroll
    for (int r = 0; r < 8; ++r) {
      int m = r + 8 * h4;
      int sq = q0 + m;
      float tot = redsum16(rsum[r]);
      float inv = (tot > 0.0f) ? 1.0f / tot : 0.0f;
      if (mrow[sq] < 0.0f) inv = 0.0f;  // is_index_masked
      float* orow = out + ((size_t)b * S_DIM + sq) * E_DIM + h * D_DIM;
      orow[n] = o0[r] * inv;
      orow[16 + n] = o1[r] * inv;
      orow[32 + n] = o2[r] * inv;
      orow[48 + n] = o3[r] * inv;
    }
  }
}

extern "C" void kernel_launch(void* const* d_in, const int* in_sizes, int n_in,
                              void* d_out, int out_size, void* d_ws,
                              size_t ws_size, hipStream_t stream) {
  const float* hidden = (const float*)d_in[0];
  const float* amask = (const float*)d_in[1];
  const float* Wq = (const float*)d_in[2];
  const float* bq = (const float*)d_in[3];
  const float* Wk = (const float*)d_in[4];
  const float* bk = (const float*)d_in[5];
  const float* Wv = (const float*)d_in[6];
  const float* bv = (const float*)d_in[7];
  float* out = (float*)d_out;

  char* ws = (char*)d_ws;
  const size_t szX = (size_t)B_DIM * S_DIM * E_DIM * sizeof(_Float16); // 12.6MB
  const size_t szW = (size_t)3 * E_DIM * E_DIM * sizeof(_Float16);
  _Float16* Xf = (_Float16*)(ws);
  _Float16* Wt = (_Float16*)(ws + szX);
  _Float16* Qf = (_Float16*)(ws + szX + szW);
  _Float16* Kf = (_Float16*)(ws + 2 * szX + szW);
  _Float16* Vf = (_Float16*)(ws + 3 * szX + szW);

  int nX = B_DIM * S_DIM * E_DIM;
  f32_to_f16_kernel<<<(nX + 255) / 256, 256, 0, stream>>>(hidden, Xf, nX);
  transpose_w_kernel<<<dim3(E_DIM * E_DIM / 256, 1, 3), 256, 0, stream>>>(
      Wq, Wk, Wv, Wt);
  qkv_gemm_kernel<<<dim3(B_DIM * S_DIM / 128, H_NUM, 3), 256, 0, stream>>>(
      Xf, Wt, bq, bk, bv, Qf, Kf, Vf);
  attn_kernel<<<dim3(C_NUM, H_NUM, B_DIM), 256, 0, stream>>>(Qf, Kf, Vf, amask,
                                                             out);
}
